// SimpleI2G_69295002353926
// MI455X (gfx1250) — matrix operand order repaired and checked by reading, not compiled, not run on hardware
//
#include <hip/hip_runtime.h>
#include <stdint.h>

// ---------------------------------------------------------------------------
// SimpleI2G (CNN encoder + ChebConv GNN deformation) for gfx1250 / MI455X.
// - Encoder convs (except Cin=1 first conv): implicit-GEMM via
//   v_wmma_f32_16x16x32_bf16 over halo-padded NHWC bf16 activations,
//   27 taps x K-steps, 64x16 output block per wave (4 accumulators).
// - Cheb GEMMs: v_wmma_f32_16x16x32_bf16, same register blocking.
// Input flattening assumption: setup_inputs() dict in insertion order:
//   0:x 1:template_x 2:edge_index 3..12:enc_ws 13..22:enc_bs
//   23+b*10: [procW,procB, defW0,defB0, defW1,defB1, defW2,defB2, outW,outB]
// ---------------------------------------------------------------------------

#define NN 8192
#define NE 49152

typedef __attribute__((ext_vector_type(16))) __bf16 v16bf;
typedef __attribute__((ext_vector_type(8)))  float  v8f;

__device__ __forceinline__ unsigned short f2bf(float f) {
  unsigned u = __float_as_uint(f);
  unsigned r = u + 0x7FFFu + ((u >> 16) & 1u);   // round-to-nearest-even
  return (unsigned short)(r >> 16);
}

// ------------------------- WMMA fragment helpers ---------------------------

union FragU { v16bf v; uint4 q[2]; };

// A fragment per ISA 7.12.2 (16-bit A 16x32): lane<16 holds K[k..k+7],K[k+16..k+23]
__device__ __forceinline__ v16bf load_frag_a(const unsigned short* p) {
  FragU f;
  f.q[0] = *(const uint4*)(p);
  f.q[1] = *(const uint4*)(p + 16);
  return f.v;
}
// B fragment: lane<16 -> col N=lane, K[k..k+7],K[k+16..k+23] (weights transposed)
__device__ __forceinline__ v16bf load_frag_b(const unsigned short* p) {
  FragU f;
  f.q[0] = *(const uint4*)(p);
  f.q[1] = *(const uint4*)(p + 16);
  return f.v;
}

#define WMMA_BF16(a, b, c) \
  __builtin_amdgcn_wmma_f32_16x16x32_bf16(false, (a), false, (b), (short)0, (c), false, false)

// ------------------------- encoder kernels ---------------------------------

// Level-0 conv1: Cin=1 (K-pad to 32 would waste 31/32 of WMMA work).
// Reads x [V^3] f32, writes halo-padded NHWC bf16 (relu), co < 16.
__global__ void conv0_direct_kernel(const float* __restrict__ x,
                                    const float* __restrict__ w,   // [16][27]
                                    const float* __restrict__ bias,
                                    unsigned short* __restrict__ outP,
                                    int V, int Kp) {
  long long idx = (long long)blockIdx.x * blockDim.x + threadIdx.x;
  long long tot = (long long)V * V * V * 16;
  if (idx >= tot) return;
  int co = (int)(idx & 15);
  long long p = idx >> 4;
  int xp = (int)(p % V);
  int yp = (int)((p / V) % V);
  int zp = (int)(p / ((long long)V * V));
  float acc = bias[co];
  const float* wp = w + co * 27;
#pragma unroll
  for (int dz = 0; dz < 3; ++dz) {
    int zz = zp + dz - 1; if (zz < 0 || zz >= V) continue;
#pragma unroll
    for (int dy = 0; dy < 3; ++dy) {
      int yy = yp + dy - 1; if (yy < 0 || yy >= V) continue;
#pragma unroll
      for (int dx = 0; dx < 3; ++dx) {
        int xx = xp + dx - 1; if (xx < 0 || xx >= V) continue;
        acc = fmaf(x[((long long)zz * V + yy) * V + xx], wp[(dz * 3 + dy) * 3 + dx], acc);
      }
    }
  }
  int P = V + 2;
  outP[(((long long)(zp + 1) * P + (yp + 1)) * P + (xp + 1)) * Kp + co] =
      f2bf(fmaxf(acc, 0.0f));
}

// Per-tap transposed weights: dst[t][n][k] = W[n][k][t], zero-padded in k.
__global__ void wtap_convert_kernel(unsigned short* __restrict__ dst,
                                    const float* __restrict__ W,   // [Cout][Cin][27]
                                    int Cin, int Cout, int Kpad) {
  long long t = (long long)blockIdx.x * blockDim.x + threadIdx.x;
  long long tot = (long long)27 * Cout * Kpad;
  if (t >= tot) return;
  int k = (int)(t % Kpad);
  int n = (int)((t / Kpad) % Cout);
  int tap = (int)(t / ((long long)Kpad * Cout));
  dst[t] = (k < Cin) ? f2bf(W[((long long)n * Cin + k) * 27 + tap]) : (unsigned short)0;
}

// Implicit-GEMM 3x3x3 conv. A = halo-padded NHWC bf16 [(V+2)^3 x Kpad],
// Wt = [27][Cout][Kpad] bf16 (K contiguous). mode 1: write NHWC f32 (relu);
// mode 0: write halo-padded NHWC bf16 (relu) with stride KpadOut.
__global__ void conv_gemm_wmma(const unsigned short* __restrict__ Ain,
                               const unsigned short* __restrict__ Wt,
                               const float* __restrict__ bias,
                               float* __restrict__ outF,
                               unsigned short* __restrict__ outP,
                               int V, int logV, int Kpad, int Cout, int KpadOut, int mode) {
  int wave = blockIdx.x * (blockDim.x >> 5) + ((int)threadIdx.x >> 5);
  int lane = (int)threadIdx.x & 31;
  int tilesN = Cout >> 4;
  long long M = (long long)V * V * V;
  int total = (int)(M >> 6) * tilesN;
  if (wave >= total) return;               // wave-uniform: EXEC stays all-ones
  int gm = wave / tilesN;
  int tn = wave % tilesN;
  int half = lane >> 4;
  int r = lane & 15;
  int P = V + 2;
  const long long bRow = (long long)(tn * 16 + r) * Kpad;
  // per-lane A base addresses for the 4 M-tiles (voxel -> padded NHWC offset)
  long long aB[4];
#pragma unroll
  for (int j = 0; j < 4; ++j) {
    int p  = gm * 64 + j * 16 + r;
    int xp = p & (V - 1);
    int yp = (p >> logV) & (V - 1);
    int zp = p >> (2 * logV);
    aB[j] = ((long long)((zp + 1) * P + (yp + 1)) * P + (xp + 1)) * Kpad;
  }
  v8f acc0 = {0.f, 0.f, 0.f, 0.f, 0.f, 0.f, 0.f, 0.f};
  v8f acc1 = acc0, acc2 = acc0, acc3 = acc0;
  for (int t = 0; t < 27; ++t) {
    int dz = t / 9, dy = (t / 3) % 3, dx = t % 3;
    long long toff = ((long long)(dz - 1) * P * P + (long long)(dy - 1) * P + (dx - 1)) * Kpad;
    const unsigned short* wt = Wt + (long long)t * Cout * Kpad + bRow;
    for (int k0 = 0; k0 < Kpad; k0 += 32) {
      int ao = k0 + half * 8;
      int bo = k0 + half * 8;
      v16bf b = load_frag_b(wt + bo);
      acc0 = WMMA_BF16(load_frag_a(Ain + aB[0] + toff + ao), b, acc0);
      acc1 = WMMA_BF16(load_frag_a(Ain + aB[1] + toff + ao), b, acc1);
      acc2 = WMMA_BF16(load_frag_a(Ain + aB[2] + toff + ao), b, acc2);
      acc3 = WMMA_BF16(load_frag_a(Ain + aB[3] + toff + ao), b, acc3);
    }
  }
  int coln = tn * 16 + r;                  // always < Cout (multiple of 16)
  float bv = bias[coln];
  v8f accs[4] = {acc0, acc1, acc2, acc3};
  if (mode == 1) {
#pragma unroll
    for (int j = 0; j < 4; ++j) {
      int rowbase = gm * 64 + j * 16 + half * 8;  // C/D: vgpr i -> voxel rowbase+i
#pragma unroll
      for (int i = 0; i < 8; ++i)
        outF[(long long)(rowbase + i) * Cout + coln] = fmaxf(accs[j][i] + bv, 0.0f);
    }
  } else {
#pragma unroll
    for (int j = 0; j < 4; ++j) {
      int rowbase = gm * 64 + j * 16 + half * 8;
#pragma unroll
      for (int i = 0; i < 8; ++i) {
        int p  = rowbase + i;
        int xp = p & (V - 1);
        int yp = (p >> logV) & (V - 1);
        int zp = p >> (2 * logV);
        outP[((long long)((zp + 1) * P + (yp + 1)) * P + (xp + 1)) * KpadOut + coln] =
            f2bf(fmaxf(accs[j][i] + bv, 0.0f));
      }
    }
  }
}

// Maxpool 2x2x2 on NHWC f32 -> halo-padded NHWC bf16 (next conv's input).
__global__ void maxpool_nhwc_pad_kernel(const float* __restrict__ in,
                                        unsigned short* __restrict__ outP,
                                        int C, int V, int Kp) {
  int Vo = V >> 1, Po = Vo + 2;
  long long idx = (long long)blockIdx.x * blockDim.x + threadIdx.x;
  long long tot = (long long)Vo * Vo * Vo * C;
  if (idx >= tot) return;
  int c = (int)(idx % C);
  long long v = idx / C;
  int x = (int)(v % Vo);
  int y = (int)((v / Vo) % Vo);
  int z = (int)(v / ((long long)Vo * Vo));
  float m = -3.402823466e38f;
#pragma unroll
  for (int dz = 0; dz < 2; ++dz)
#pragma unroll
    for (int dy = 0; dy < 2; ++dy)
#pragma unroll
      for (int dx = 0; dx < 2; ++dx)
        m = fmaxf(m, in[(((long long)(2 * z + dz) * V + (2 * y + dy)) * V + (2 * x + dx)) * C + c]);
  outP[(((long long)(z + 1) * Po + (y + 1)) * Po + (x + 1)) * Kp + c] = f2bf(m);
}

// ------------------------- graph-weight kernels ----------------------------

__global__ void deg_kernel(float* __restrict__ deg, const int* __restrict__ row, int E) {
  int e = blockIdx.x * blockDim.x + threadIdx.x;
  if (e < E) atomicAdd(&deg[row[e]], 1.0f);
}

__global__ void dinv_kernel(float* __restrict__ deg, int N) {
  int i = blockIdx.x * blockDim.x + threadIdx.x;
  if (i < N) {
    float d = deg[i];
    deg[i] = (d > 0.0f) ? rsqrtf(fmaxf(d, 1.0f)) : 0.0f;  // in place -> dinv
  }
}

__global__ void normw_kernel(float* __restrict__ nw, const float* __restrict__ dinv,
                             const int* __restrict__ row, const int* __restrict__ col, int E) {
  int e = blockIdx.x * blockDim.x + threadIdx.x;
  if (e < E) nw[e] = -dinv[row[e]] * dinv[col[e]];
}

// ------------------------- cheb propagation --------------------------------

__global__ void scatter_prop_kernel(float* __restrict__ dst, const float* __restrict__ src,
                                    const int* __restrict__ row, const int* __restrict__ col,
                                    const float* __restrict__ nw, int E, int C, float scale) {
  long long t = (long long)blockIdx.x * blockDim.x + threadIdx.x;
  if (t >= (long long)E * C) return;
  int e = (int)(t / C);
  int c = (int)(t % C);
  atomicAdd(&dst[(long long)col[e] * C + c],
            scale * nw[e] * src[(long long)row[e] * C + c]);
}

__global__ void neg_copy_kernel(float* __restrict__ dst, const float* __restrict__ src, long long n) {
  long long i = (long long)blockIdx.x * blockDim.x + threadIdx.x;
  if (i < n) dst[i] = -src[i];
}

// ------------------------- bf16 staging (GNN) ------------------------------

__global__ void act_convert_kernel(unsigned short* __restrict__ dst,
                                   const float* __restrict__ src,
                                   int N, int Kex, int Kpad) {
  long long t = (long long)blockIdx.x * blockDim.x + threadIdx.x;
  if (t >= (long long)N * Kpad) return;
  int n = (int)(t / Kpad);
  int k = (int)(t % Kpad);
  dst[t] = (k < Kex) ? f2bf(src[(long long)n * Kex + k]) : (unsigned short)0;
}

// W: [Kex x Nex] f32 -> dst: [Npad x Kpad] bf16 transposed (K contiguous)
__global__ void wt_convert_kernel(unsigned short* __restrict__ dst,
                                  const float* __restrict__ W,
                                  int Kex, int Nex, int Kpad, int Npad) {
  long long t = (long long)blockIdx.x * blockDim.x + threadIdx.x;
  if (t >= (long long)Npad * Kpad) return;
  int n = (int)(t / Kpad);
  int k = (int)(t % Kpad);
  dst[t] = (k < Kex && n < Nex) ? f2bf(W[(long long)k * Nex + n]) : (unsigned short)0;
}

// ------------------------- Cheb WMMA GEMM ----------------------------------

// C[M x Nex] = A0@W0 + A1@W1 + A2@W2 + bias  (bf16 in, f32 out).
// 64x16 block per wave: every B fragment reused by 4 WMMAs.
__global__ void cheb_gemm_wmma(float* __restrict__ Cmat,
                               const unsigned short* __restrict__ A0,
                               const unsigned short* __restrict__ A1,
                               const unsigned short* __restrict__ A2,
                               const unsigned short* __restrict__ B0,
                               const unsigned short* __restrict__ B1,
                               const unsigned short* __restrict__ B2,
                               const float* __restrict__ bias,
                               int M, int Nex, int Kpad, int Npad) {
  int wave = blockIdx.x * (blockDim.x >> 5) + ((int)threadIdx.x >> 5);
  int lane = (int)threadIdx.x & 31;
  int tilesN = Npad >> 4;
  int total = (M >> 6) * tilesN;
  if (wave >= total) return;               // wave-uniform: EXEC stays all-ones
  int gm = wave / tilesN;
  int tn = wave % tilesN;
  int half = lane >> 4;
  int r = lane & 15;
  const long long aBase = (long long)(gm * 64 + r) * Kpad;
  const long long aStep = (long long)16 * Kpad;
  const long long bRow  = (long long)(tn * 16 + r) * Kpad;
  v8f acc0 = {0.f, 0.f, 0.f, 0.f, 0.f, 0.f, 0.f, 0.f};
  v8f acc1 = acc0, acc2 = acc0, acc3 = acc0;
  for (int k0 = 0; k0 < Kpad; k0 += 32) {
    const int ao = k0 + half * 8;
    const int bo = k0 + half * 8;
    v16bf b0 = load_frag_b(B0 + bRow + bo);
    v16bf b1 = load_frag_b(B1 + bRow + bo);
    v16bf b2 = load_frag_b(B2 + bRow + bo);
    const unsigned short* pa0 = A0 + aBase + ao;
    const unsigned short* pa1 = A1 + aBase + ao;
    const unsigned short* pa2 = A2 + aBase + ao;
    acc0 = WMMA_BF16(load_frag_a(pa0), b0, acc0);
    acc0 = WMMA_BF16(load_frag_a(pa1), b1, acc0);
    acc0 = WMMA_BF16(load_frag_a(pa2), b2, acc0);
    acc1 = WMMA_BF16(load_frag_a(pa0 + aStep), b0, acc1);
    acc1 = WMMA_BF16(load_frag_a(pa1 + aStep), b1, acc1);
    acc1 = WMMA_BF16(load_frag_a(pa2 + aStep), b2, acc1);
    acc2 = WMMA_BF16(load_frag_a(pa0 + 2 * aStep), b0, acc2);
    acc2 = WMMA_BF16(load_frag_a(pa1 + 2 * aStep), b1, acc2);
    acc2 = WMMA_BF16(load_frag_a(pa2 + 2 * aStep), b2, acc2);
    acc3 = WMMA_BF16(load_frag_a(pa0 + 3 * aStep), b0, acc3);
    acc3 = WMMA_BF16(load_frag_a(pa1 + 3 * aStep), b1, acc3);
    acc3 = WMMA_BF16(load_frag_a(pa2 + 3 * aStep), b2, acc3);
  }
  int coln = tn * 16 + r;
  if (coln >= Nex) return;                 // padded-N tiles write nothing
  float bv = bias[coln];
  v8f accs[4] = {acc0, acc1, acc2, acc3};
#pragma unroll
  for (int j = 0; j < 4; ++j) {
    int rowbase = gm * 64 + j * 16 + half * 8;   // C/D: vgpr i -> M = rowbase+i
#pragma unroll
    for (int i = 0; i < 8; ++i)
      Cmat[(long long)(rowbase + i) * Nex + coln] = accs[j][i] + bv;
  }
}

// ------------------------- instance norm + relu ----------------------------

__global__ void inorm_relu_kernel(float* __restrict__ X, int N, int C) {
  int c = blockIdx.x;
  __shared__ float ssum[256];
  __shared__ float ssq[256];
  float a = 0.f, b = 0.f;
  for (int n = (int)threadIdx.x; n < N; n += (int)blockDim.x) {
    float v = X[(long long)n * C + c];
    a += v; b += v * v;
  }
  ssum[threadIdx.x] = a; ssq[threadIdx.x] = b;
  __syncthreads();
  for (int s = (int)blockDim.x >> 1; s > 0; s >>= 1) {
    if ((int)threadIdx.x < s) {
      ssum[threadIdx.x] += ssum[threadIdx.x + s];
      ssq[threadIdx.x]  += ssq[threadIdx.x + s];
    }
    __syncthreads();
  }
  float mean = ssum[0] / (float)N;
  float var  = ssq[0] / (float)N - mean * mean;
  float inv  = rsqrtf(var + 1e-5f);
  for (int n = (int)threadIdx.x; n < N; n += (int)blockDim.x) {
    float v = (X[(long long)n * C + c] - mean) * inv;
    X[(long long)n * C + c] = fmaxf(v, 0.0f);
  }
}

// ------------------------- trilinear sampling (NHWC fmaps) -----------------

__global__ void trilinear_kernel(const float* __restrict__ fmap, int C, int S,
                                 const float* __restrict__ coords,  // [N,3]
                                 float* __restrict__ dst, int dstStride, int dstOfs, int N) {
  long long t = (long long)blockIdx.x * blockDim.x + threadIdx.x;
  if (t >= (long long)N * C) return;
  int n = (int)(t / C);
  int c = (int)(t % C);
  float s1 = (float)(S - 1);
  float gx = fminf(fmaxf((coords[n * 3 + 0] + 1.f) * 0.5f * s1, 0.f), s1);
  float gy = fminf(fmaxf((coords[n * 3 + 1] + 1.f) * 0.5f * s1, 0.f), s1);
  float gz = fminf(fmaxf((coords[n * 3 + 2] + 1.f) * 0.5f * s1, 0.f), s1);
  int x0 = (int)floorf(gx), y0 = (int)floorf(gy), z0 = (int)floorf(gz);
  int x1 = min(x0 + 1, S - 1), y1 = min(y0 + 1, S - 1), z1 = min(z0 + 1, S - 1);
  float wx = gx - (float)x0, wy = gy - (float)y0, wz = gz - (float)z0;
  auto G = [&](int zi, int yi, int xi) {
    return fmap[(((long long)zi * S + yi) * S + xi) * C + c];
  };
  float c00 = G(z0, y0, x0) * (1.f - wx) + G(z0, y0, x1) * wx;
  float c01 = G(z0, y1, x0) * (1.f - wx) + G(z0, y1, x1) * wx;
  float c10 = G(z1, y0, x0) * (1.f - wx) + G(z1, y0, x1) * wx;
  float c11 = G(z1, y1, x0) * (1.f - wx) + G(z1, y1, x1) * wx;
  float v = (c00 * (1.f - wy) + c01 * wy) * (1.f - wz) +
            (c10 * (1.f - wy) + c11 * wy) * wz;
  dst[(long long)n * dstStride + dstOfs + c] = v;
}

// ------------------------- small utility kernels ---------------------------

__global__ void copy_f32_kernel(float* __restrict__ dst, const float* __restrict__ src, long long n) {
  long long i = (long long)blockIdx.x * blockDim.x + threadIdx.x;
  if (i < n) dst[i] = src[i];
}

__global__ void copy_cols_kernel(float* __restrict__ dst, int dstStride,
                                 const float* __restrict__ src, int srcC, int N) {
  long long t = (long long)blockIdx.x * blockDim.x + threadIdx.x;
  if (t >= (long long)N * srcC) return;
  int n = (int)(t / srcC), c = (int)(t % srcC);
  dst[(long long)n * dstStride + c] = src[(long long)n * srcC + c];
}

__global__ void out_update_kernel(float* __restrict__ o, const float* __restrict__ d,
                                  float* __restrict__ dOut, int n) {
  int i = blockIdx.x * blockDim.x + threadIdx.x;
  if (i < n) {
    float v = o[i] + d[i];
    o[i] = v;
    dOut[i] = v;
  }
}

// ------------------------- host side ---------------------------------------

static inline int gblk(long long n, int b) { return (int)((n + b - 1) / b); }
static inline char* carve(char*& p, size_t bytes) {
  char* r = p;
  p += (bytes + 255) & ~(size_t)255;
  return r;
}
static inline int al32(int x) { return (x + 31) & ~31; }
static inline int al16(int x) { return (x + 15) & ~15; }

struct ChebCtx {
  hipStream_t s;
  const int* row;
  const int* col;
  float* nw;
  float* FB[4];
  unsigned short* AB[3];
  unsigned short* WT[3];
};

// Runs one ChebConv layer (K=3) from FB[curIdx] (Cin) -> returns result index.
static int run_cheb(ChebCtx& c, int curIdx, int Cin, int Cout,
                    const float* W, const float* bias, bool doNorm) {
  float* cur = c.FB[curIdx];
  float* t1  = c.FB[(curIdx + 1) & 3];
  float* t2  = c.FB[(curIdx + 2) & 3];
  int resIdx = (curIdx + 3) & 3;
  float* res = c.FB[resIdx];
  int Kpad = al32(Cin), Npad = al16(Cout);

  hipMemsetAsync(t1, 0, (size_t)NN * Cin * sizeof(float), c.s);
  scatter_prop_kernel<<<gblk((long long)NE * Cin, 256), 256, 0, c.s>>>(
      t1, cur, c.row, c.col, c.nw, NE, Cin, 1.0f);
  neg_copy_kernel<<<gblk((long long)NN * Cin, 256), 256, 0, c.s>>>(
      t2, cur, (long long)NN * Cin);
  scatter_prop_kernel<<<gblk((long long)NE * Cin, 256), 256, 0, c.s>>>(
      t2, t1, c.row, c.col, c.nw, NE, Cin, 2.0f);

  const float* T[3] = {cur, t1, t2};
  for (int k = 0; k < 3; ++k) {
    act_convert_kernel<<<gblk((long long)NN * Kpad, 256), 256, 0, c.s>>>(
        c.AB[k], T[k], NN, Cin, Kpad);
    wt_convert_kernel<<<gblk((long long)Npad * Kpad, 256), 256, 0, c.s>>>(
        c.WT[k], W + (long long)k * Cin * Cout, Cin, Cout, Kpad, Npad);
  }

  int total = (NN / 64) * (Npad / 16);
  cheb_gemm_wmma<<<gblk(total, 8), 256, 0, c.s>>>(
      res, c.AB[0], c.AB[1], c.AB[2], c.WT[0], c.WT[1], c.WT[2],
      bias, NN, Cout, Kpad, Npad);

  if (doNorm)
    inorm_relu_kernel<<<Cout, 256, 0, c.s>>>(res, NN, Cout);
  return resIdx;
}

extern "C" void kernel_launch(void* const* d_in, const int* in_sizes, int n_in,
                              void* d_out, int out_size, void* d_ws, size_t ws_size,
                              hipStream_t stream) {
  (void)in_sizes; (void)n_in; (void)out_size; (void)ws_size;
  const float* x    = (const float*)d_in[0];
  const float* tmpl = (const float*)d_in[1];
  const int*   eidx = (const int*)d_in[2];
  const int* row = eidx;
  const int* col = eidx + NE;
  const float* encW[10];
  const float* encB[10];
  for (int i = 0; i < 10; ++i) {
    encW[i] = (const float*)d_in[3 + i];
    encB[i] = (const float*)d_in[13 + i];
  }
  const float* blkW[3][5];
  const float* blkB[3][5];
  for (int b = 0; b < 3; ++b)
    for (int l = 0; l < 5; ++l) {
      blkW[b][l] = (const float*)d_in[23 + b * 10 + l * 2];
      blkB[b][l] = (const float*)d_in[23 + b * 10 + l * 2 + 1];
    }

  // ---- workspace layout ----
  static const int CNNF[5] = {16, 32, 64, 128, 256};
  static const int VS[5]   = {64, 32, 16, 8, 4};
  static const int LOGV[5] = {6, 5, 4, 3, 2};
  char* p = (char*)d_ws;
  float* fmap[5];                        // NHWC f32: [V^3][Cf]
  for (int l = 0; l < 5; ++l) {
    long long vol = (long long)VS[l] * VS[l] * VS[l];
    fmap[l] = (float*)carve(p, (size_t)CNNF[l] * vol * sizeof(float));
  }
  // halo-padded NHWC bf16 activation buffers (ping/pong)
  unsigned short* padA = (unsigned short*)carve(p, (size_t)66 * 66 * 66 * 32 * 2);
  unsigned short* padB = (unsigned short*)carve(p, (size_t)34 * 34 * 34 * 32 * 2);
  unsigned short* WTc  = (unsigned short*)carve(p, (size_t)27 * 256 * 256 * 2);
  ChebCtx ctx;
  ctx.s = stream; ctx.row = row; ctx.col = col;
  for (int i = 0; i < 4; ++i)
    ctx.FB[i] = (float*)carve(p, (size_t)NN * 448 * sizeof(float));
  for (int i = 0; i < 3; ++i)
    ctx.AB[i] = (unsigned short*)carve(p, (size_t)NN * 448 * sizeof(unsigned short));
  for (int i = 0; i < 3; ++i)
    ctx.WT[i] = (unsigned short*)carve(p, (size_t)448 * 448 * sizeof(unsigned short));
  float* gbuf = (float*)carve(p, (size_t)NN * 384 * sizeof(float));
  float* outv = (float*)carve(p, (size_t)NN * 3 * sizeof(float));
  float* deg  = (float*)carve(p, (size_t)NN * sizeof(float));
  ctx.nw      = (float*)carve(p, (size_t)NE * sizeof(float));

  // ---- encoder ----
  {
    // level 0: conv1 direct (Cin=1), conv2 implicit-GEMM
    int V = 64, Pd = 66, Kp = 32;
    hipMemsetAsync(padA, 0, (size_t)Pd * Pd * Pd * Kp * 2, stream);
    conv0_direct_kernel<<<gblk((long long)V * V * V * 16, 256), 256, 0, stream>>>(
        x, encW[0], encB[0], padA, V, Kp);
    wtap_convert_kernel<<<gblk((long long)27 * 16 * Kp, 256), 256, 0, stream>>>(
        WTc, encW[1], 16, 16, Kp);
    conv_gemm_wmma<<<gblk(((long long)V * V * V / 64) * (16 / 16), 8), 256, 0, stream>>>(
        padA, WTc, encB[1], fmap[0], nullptr, V, 6, Kp, 16, 0, 1);
    hipMemsetAsync(padB, 0, (size_t)34 * 34 * 34 * 32 * 2, stream);
    maxpool_nhwc_pad_kernel<<<gblk((long long)32 * 32 * 32 * 16, 256), 256, 0, stream>>>(
        fmap[0], padB, 16, V, 32);
  }
  for (int l = 1; l < 5; ++l) {
    int Cin = CNNF[l - 1], Cf = CNNF[l], V = VS[l], Pd = V + 2;
    int Kin = al32(Cin), Kout = al32(Cf);
    long long vol = (long long)V * V * V;
    // conv1: padB (Cin, Kin) -> padA (Cf, Kout, halo-padded bf16)
    hipMemsetAsync(padA, 0, (size_t)Pd * Pd * Pd * Kout * 2, stream);
    wtap_convert_kernel<<<gblk((long long)27 * Cf * Kin, 256), 256, 0, stream>>>(
        WTc, encW[2 * l], Cin, Cf, Kin);
    conv_gemm_wmma<<<gblk((vol / 64) * (Cf / 16), 8), 256, 0, stream>>>(
        padB, WTc, encB[2 * l], nullptr, padA, V, LOGV[l], Kin, Cf, Kout, 0);
    // conv2: padA -> fmap[l] (NHWC f32)
    wtap_convert_kernel<<<gblk((long long)27 * Cf * Kout, 256), 256, 0, stream>>>(
        WTc, encW[2 * l + 1], Cf, Cf, Kout);
    conv_gemm_wmma<<<gblk((vol / 64) * (Cf / 16), 8), 256, 0, stream>>>(
        padA, WTc, encB[2 * l + 1], fmap[l], nullptr, V, LOGV[l], Kout, Cf, 0, 1);
    if (l < 4) {
      int Vo = V >> 1, Po = Vo + 2;
      hipMemsetAsync(padB, 0, (size_t)Po * Po * Po * Kout * 2, stream);
      maxpool_nhwc_pad_kernel<<<gblk((long long)Vo * Vo * Vo * Cf, 256), 256, 0, stream>>>(
          fmap[l], padB, Cf, V, Kout);
    }
  }

  // ---- graph normalization weights ----
  hipMemsetAsync(deg, 0, (size_t)NN * sizeof(float), stream);
  deg_kernel<<<gblk(NE, 256), 256, 0, stream>>>(deg, row, NE);
  dinv_kernel<<<gblk(NN, 256), 256, 0, stream>>>(deg, NN);
  normw_kernel<<<gblk(NE, 256), 256, 0, stream>>>(ctx.nw, deg, row, col, NE);

  // ---- GNN blocks ----
  copy_f32_kernel<<<gblk((long long)NN * 3, 256), 256, 0, stream>>>(
      outv, tmpl, (long long)NN * 3);

  struct BCfg { int id0, id1, c0, c1, gch, f0, f1; };
  const BCfg cfg[3] = {
      {3, 4, 128, 256, 3,   384, 288},
      {1, 2, 32,  64,  288, 144, 96},
      {0, 1, 16,  32,  96,  64,  32},
  };
  const float* graphSrc = tmpl;
  int graphC = 3;

  for (int b = 0; b < 3; ++b) {
    const BCfg& B = cfg[b];
    int Cin0 = B.gch + B.c0 + B.c1;
    int curIdx = 0;
    // concat [graph, proj0, proj1] into FB[0] with stride Cin0
    copy_cols_kernel<<<gblk((long long)NN * graphC, 256), 256, 0, stream>>>(
        ctx.FB[0], Cin0, graphSrc, graphC, NN);
    trilinear_kernel<<<gblk((long long)NN * B.c0, 256), 256, 0, stream>>>(
        fmap[B.id0], B.c0, VS[B.id0], outv, ctx.FB[0], Cin0, graphC, NN);
    trilinear_kernel<<<gblk((long long)NN * B.c1, 256), 256, 0, stream>>>(
        fmap[B.id1], B.c1, VS[B.id1], outv, ctx.FB[0], Cin0, graphC + B.c0, NN);
    // process (1) + deform (3) layers, each with inorm+relu
    int lc = Cin0;
    for (int l = 0; l < 4; ++l) {
      int co = (l == 0) ? B.f0 : B.f1;
      curIdx = run_cheb(ctx, curIdx, lc, co, blkW[b][l], blkB[b][l], true);
      lc = co;
    }
    // out layer (linear), residual update, write to d_out[b]
    int resIdx = run_cheb(ctx, curIdx, lc, 3, blkW[b][4], blkB[b][4], false);
    out_update_kernel<<<gblk(NN * 3, 256), 256, 0, stream>>>(
        outv, ctx.FB[resIdx], (float*)d_out + (long long)b * NN * 3, NN * 3);
    // graph = h
    copy_f32_kernel<<<gblk((long long)NN * lc, 256), 256, 0, stream>>>(
        gbuf, ctx.FB[curIdx], (long long)NN * lc);
    graphSrc = gbuf;
    graphC = lc;
  }
}
